// PopLSTM_257698038289
// MI455X (gfx1250) — matrix operand order repaired and checked by reading, not compiled, not run on hardware
//
#include <hip/hip_runtime.h>
#include <hip/hip_bf16.h>
#include <stdint.h>

// ---------------------------------------------------------------------------
// Problem constants (PopLSTM): B=128, T=512, D=512, U=512, NPOP=32
// ---------------------------------------------------------------------------
#define PB   128
#define PT   512
#define PD   512
#define PU   512
#define P4U  2048
#define NPOP 32

typedef __bf16 bf16;
typedef bf16  v16bf __attribute__((ext_vector_type(16)));
typedef float v8f   __attribute__((ext_vector_type(8)));

union FragU { v16bf v; uint4 q[2]; };

// A fragment (16x32 bf16, row-major source, leading dim ld):
//   lane L: m = L&15 row; hsel = L>>4 selects K sub-halves.
//   halves 0..7  <-> K = k0 + 8*hsel + j
//   halves 8..15 <-> K = k0 + 16 + 8*hsel + j
__device__ __forceinline__ v16bf load_a_frag(const bf16* A, int ld, int m,
                                             int hsel, int k0) {
    FragU f;
    const bf16* base = A + (size_t)m * ld + k0 + 8 * hsel;
    f.q[0] = *reinterpret_cast<const uint4*>(base);
    f.q[1] = *reinterpret_cast<const uint4*>(base + 16);
    return f.v;
}

// B fragment (32x16 bf16) from K-transposed weights Bt[n, k] (ld = K):
//   lane L: n = L&15 column; kg = L>>4; halves j=0..7 <-> K = k0 + 8*kg + j, j=8..15 <-> K = k0 + 16 + 8*kg + (j-8)
__device__ __forceinline__ v16bf load_b_frag(const bf16* Bt, int ld, int n,
                                             int kg, int k0) {
    FragU f;
    const bf16* base = Bt + (size_t)n * ld + k0 + 8 * kg;
    f.q[0] = *reinterpret_cast<const uint4*>(base);
    f.q[1] = *reinterpret_cast<const uint4*>(base + 16);
    return f.v;
}

// ---------------------------------------------------------------------------
// Setup kernels
// ---------------------------------------------------------------------------
__global__ void poplstm_cvt_f32_bf16(const float* __restrict__ in,
                                     bf16* __restrict__ out, int n) {
    int i = blockIdx.x * blockDim.x + threadIdx.x;
    if (i < n) out[i] = (bf16)in[i];
}

// in: [K, N] f32 row-major  ->  out: [N, K] bf16 (K-contiguous)
__global__ void poplstm_transpose_to_bf16(const float* __restrict__ in,
                                          bf16* __restrict__ out, int K, int N) {
    int i = blockIdx.x * blockDim.x + threadIdx.x;
    if (i < K * N) {
        int k = i / N, n = i - k * N;
        out[(size_t)n * K + k] = (bf16)in[i];
    }
}

__global__ void poplstm_init_state(const float* __restrict__ h0,
                                   const float* __restrict__ c0,
                                   bf16* __restrict__ hbuf,
                                   float* __restrict__ cws, int n) {
    int i = blockIdx.x * blockDim.x + threadIdx.x;
    if (i < n) { hbuf[i] = (bf16)h0[i]; cws[i] = c0[i]; }
}

// ---------------------------------------------------------------------------
// Phase 1: xz[b,t,:] = x[b,t,:] @ W + bias        (M = B*T = 65536, N = 2048,
// K = 512).  Block = 256 threads = 8 waves; wave w owns the 16x16 tile at
// (m0 = 16*blockIdx.y, n0 = 128*blockIdx.x + 16*w).
// ---------------------------------------------------------------------------
__global__ __launch_bounds__(256)
void poplstm_gemm_x(const bf16* __restrict__ xbf,  // [B*T, D] bf16
                    const bf16* __restrict__ Wt,   // [4U, D]  bf16 (KT)
                    const float* __restrict__ bias,
                    float* __restrict__ xz) {      // [B*T, 4U] f32
    const int lane = threadIdx.x & 31;
    const int wave = threadIdx.x >> 5;
    const int n0 = blockIdx.x * 128 + wave * 16;
    const int m0 = blockIdx.y * 16;
    const int ml = lane & 15;
    const int hsel = lane >> 4;

    v8f acc = {};
#pragma unroll 4
    for (int k0 = 0; k0 < PD; k0 += 32) {
        __builtin_prefetch(Wt + (size_t)(n0 + ml) * PD + k0 + 64, 0, 1);
        v16bf a = load_a_frag(xbf, PD, m0 + ml, hsel, k0);
        v16bf b = load_b_frag(Wt, PD, n0 + ml, hsel, k0);
        acc = __builtin_amdgcn_wmma_f32_16x16x32_bf16(
            false, a, false, b, (short)0, acc, false, false);
    }

    const int n = n0 + ml;
    const float bv = bias[n];
#pragma unroll
    for (int r = 0; r < 8; ++r) {
        int m = m0 + r + 8 * hsel;
        xz[(size_t)m * P4U + n] = acc[r] + bv;
    }
}

// ---------------------------------------------------------------------------
// Phase 2: one fused kernel per timestep.
// Grid: 256 blocks; block = (b-tile, u-tile) of 16x16. 128 threads = 4 waves,
// wave g computes the recurrent-GEMM tile for gate g (columns g*U + u0..+15),
// adds the precomputed xz slice, parks z in LDS; then all threads do the
// population-threshold gate math and write c_t / h_t.
// ---------------------------------------------------------------------------
__global__ __launch_bounds__(128)
void poplstm_step(const bf16* __restrict__ hin,   // [B, U] bf16 (h_{t-1})
                  bf16* __restrict__ hout,        // [B, U] bf16 (h_t)
                  float* __restrict__ cws,        // [B, U] f32
                  const float* __restrict__ xz,   // [B, T, 4U] f32
                  const bf16* __restrict__ Rt,    // [4U, U] bf16 (KT)
                  const float* __restrict__ i_th,
                  const float* __restrict__ f_th,
                  const float* __restrict__ c_tp,
                  const float* __restrict__ c_tn,
                  const float* __restrict__ o_th,
                  const float* __restrict__ h_tp,
                  const float* __restrict__ h_tn,
                  float* __restrict__ out,        // [B, T, U] f32
                  int t) {
    __shared__ float zbuf[4][16][16];

    const int lane = threadIdx.x & 31;
    const int g = threadIdx.x >> 5;                 // gate 0..3
    const int ut = blockIdx.x & 31;
    const int bt = blockIdx.x >> 5;
    const int b0 = bt * 16;
    const int u0 = ut * 16;
    const int ml = lane & 15;
    const int hsel = lane >> 4;
    const int ncol = g * PU + u0 + ml;              // column in [0, 4U)

    v8f acc = {};
#pragma unroll 4
    for (int k0 = 0; k0 < PU; k0 += 32) {
        __builtin_prefetch(Rt + (size_t)ncol * PU + k0 + 64, 0, 1);
        v16bf a = load_a_frag(hin, PU, b0 + ml, hsel, k0);
        v16bf b = load_b_frag(Rt, PU, ncol, hsel, k0);
        acc = __builtin_amdgcn_wmma_f32_16x16x32_bf16(
            false, a, false, b, (short)0, acc, false, false);
    }

    // z = h@R + (x@W + bias); accumulator: vgpr r -> m = r + 8*hsel, n = ml
#pragma unroll
    for (int r = 0; r < 8; ++r) {
        int m = r + 8 * hsel;
        size_t idx = ((size_t)(b0 + m) * PT + t) * P4U + ncol;
        zbuf[g][m][ml] = acc[r] + xz[idx];
    }
    __syncthreads();

    // Pointwise population-coded gates: 256 elements / 128 threads.
#pragma unroll
    for (int e = threadIdx.x; e < 256; e += 128) {
        const int m = e >> 4, nn = e & 15;
        const int b = b0 + m, u = u0 + nn;
        const float z0 = zbuf[0][m][nn];
        const float z1 = zbuf[1][m][nn];
        const float z2 = zbuf[2][m][nn];
        const float z3 = zbuf[3][m][nn];
        const int base = u * NPOP;

        int ci = 0, cf = 0, cc = 0, co = 0;
#pragma unroll 8
        for (int p = 0; p < NPOP; ++p) {
            ci += (z0 > i_th[base + p]);
            cf += (z1 > f_th[base + p]);
            cc += (z2 > c_tp[base + p]);
            cc -= (z2 < c_tn[base + p]);
            co += (z3 > o_th[base + p]);
        }
        const float inv = 1.0f / (float)NPOP;
        const float gi = ci * inv, gf = cf * inv, gc = cc * inv, go = co * inv;

        const size_t su = (size_t)b * PU + u;
        const float cnew = gf * cws[su] + gi * gc;

        int ch = 0;
#pragma unroll 8
        for (int p = 0; p < NPOP; ++p) {
            ch += (cnew > h_tp[base + p]);
            ch -= (cnew < h_tn[base + p]);
        }
        const float h = go * (ch * inv);

        cws[su] = cnew;
        out[((size_t)b * PT + t) * PU + u] = h;
        hout[su] = (bf16)h;
    }
}

// ---------------------------------------------------------------------------
// Host side
// ---------------------------------------------------------------------------
extern "C" void kernel_launch(void* const* d_in, const int* in_sizes, int n_in,
                              void* d_out, int out_size, void* d_ws, size_t ws_size,
                              hipStream_t stream) {
    const float* x    = (const float*)d_in[0];
    const float* h0   = (const float*)d_in[1];
    const float* c0   = (const float*)d_in[2];
    const float* W    = (const float*)d_in[3];   // [D, 4U]
    const float* R    = (const float*)d_in[4];   // [U, 4U]
    const float* bias = (const float*)d_in[5];
    const float* i_th = (const float*)d_in[6];
    const float* f_th = (const float*)d_in[7];
    const float* c_tp = (const float*)d_in[8];
    const float* c_tn = (const float*)d_in[9];
    const float* o_th = (const float*)d_in[10];
    const float* h_tp = (const float*)d_in[11];
    const float* h_tn = (const float*)d_in[12];
    float* out = (float*)d_out;

    // Workspace layout
    char* ws = (char*)d_ws;
    const size_t XZ_BYTES   = (size_t)PB * PT * P4U * sizeof(float); // 512 MB
    const size_t XBF_BYTES  = (size_t)PB * PT * PD * sizeof(bf16);   //  64 MB
    const size_t WT_BYTES   = (size_t)P4U * PD * sizeof(bf16);       //   2 MB
    const size_t RT_BYTES   = (size_t)P4U * PU * sizeof(bf16);       //   2 MB
    const size_t HB_BYTES   = (size_t)PB * PU * sizeof(bf16);        // 128 KB

    float* xz  = (float*)(ws);
    bf16*  xbf = (bf16*)(ws + XZ_BYTES);
    bf16*  Wt  = (bf16*)(ws + XZ_BYTES + XBF_BYTES);
    bf16*  Rt  = (bf16*)(ws + XZ_BYTES + XBF_BYTES + WT_BYTES);
    bf16*  hb0 = (bf16*)(ws + XZ_BYTES + XBF_BYTES + WT_BYTES + RT_BYTES);
    bf16*  hb1 = (bf16*)(ws + XZ_BYTES + XBF_BYTES + WT_BYTES + RT_BYTES + HB_BYTES);
    float* cws = (float*)(ws + XZ_BYTES + XBF_BYTES + WT_BYTES + RT_BYTES + 2 * HB_BYTES);

    // 1) x -> bf16
    {
        int n = PB * PT * PD;
        poplstm_cvt_f32_bf16<<<(n + 255) / 256, 256, 0, stream>>>(x, xbf, n);
    }
    // 2) W, R -> K-transposed bf16
    {
        int n = PD * P4U;
        poplstm_transpose_to_bf16<<<(n + 255) / 256, 256, 0, stream>>>(W, Wt, PD, P4U);
        poplstm_transpose_to_bf16<<<(n + 255) / 256, 256, 0, stream>>>(R, Rt, PU, P4U);
    }
    // 3) state init
    {
        int n = PB * PU;
        poplstm_init_state<<<(n + 255) / 256, 256, 0, stream>>>(h0, c0, hb0, cws, n);
    }
    // 4) big bias-fused GEMM: xz = x @ W + bias
    {
        dim3 grid(P4U / 128, (PB * PT) / 16);   // 16 x 4096
        poplstm_gemm_x<<<grid, 256, 0, stream>>>(xbf, Wt, bias, xz);
    }
    // 5) sequential fused steps (h double-buffered)
    for (int t = 0; t < PT; ++t) {
        bf16* hin  = (t & 1) ? hb1 : hb0;
        bf16* hout = (t & 1) ? hb0 : hb1;
        poplstm_step<<<(PB / 16) * (PU / 16), 128, 0, stream>>>(
            hin, hout, cws, xz, Rt,
            i_th, f_th, c_tp, c_tn, o_th, h_tp, h_tn,
            out, t);
    }
}
